// EncoderLayer_26285199851626
// MI455X (gfx1250) — matrix operand order repaired and checked by reading, not compiled, not run on hardware
//
#include <hip/hip_runtime.h>
#include <hip/hip_bf16.h>

// ---------------------------------------------------------------------------
// Types for CDNA5 WMMA (gfx1250, wave32)
// ---------------------------------------------------------------------------
typedef _Float16 half_t;
typedef __attribute__((ext_vector_type(16))) _Float16 v16h;
typedef __attribute__((ext_vector_type(8)))  float    v8f;
typedef __attribute__((ext_vector_type(4)))  int      v4i;

#define WMMA_F32_F16(a, b, c) \
  __builtin_amdgcn_wmma_f32_16x16x32_f16(false, (a), false, (b), (short)0, (c), false, false)

// Async global->LDS staging (GLOBAL_LOAD_ASYNC_TO_LDS_B128, ASYNCcnt-tracked)
#if defined(__has_builtin)
#if __has_builtin(__builtin_amdgcn_global_load_async_to_lds_b128)
#define HAVE_ASYNC_LDS 1
#endif
#endif

__device__ __forceinline__ void stage16B(const half_t* g, half_t* l) {
#ifdef HAVE_ASYNC_LDS
  __builtin_amdgcn_global_load_async_to_lds_b128(
      (__attribute__((address_space(1))) v4i*)g,
      (__attribute__((address_space(3))) v4i*)l, 0, 0);
#else
#pragma unroll
  for (int i = 0; i < 8; ++i) l[i] = g[i];
#endif
}

__device__ __forceinline__ void wait_async() {
#ifdef HAVE_ASYNC_LDS
  asm volatile("s_wait_asynccnt 0x0" ::: "memory");
#endif
}

// Problem constants (match reference)
#define BATCH 2
#define SEQ   1024
#define HID_  1024
#define HEADS 16
#define HDIM  64
#define PF_   4096
#define ROWS  (BATCH * SEQ)   // 2048

// ---------------------------------------------------------------------------
// A-fragment loader: 16x32 f16 A tile, row-major source with leading dim lda.
// Contiguous K order (matches B frags): lanes 0-15 -> M=lane, K in {0..15};
// lanes 16-31 -> M=lane-16, K in {16..31}; packed 2 halves per VGPR.
// ---------------------------------------------------------------------------
__device__ __forceinline__ v16h load_a_frag16(const half_t* A, int lda, int lane) {
  const int r = lane & 15;
  const int h = lane >> 4;
  const half_t* row = A + (size_t)r * lda;
  v16h f;
#pragma unroll
  for (int p = 0; p < 8; ++p) {
    int k = h * 16 + 2 * p;
    f[2 * p]     = row[k];
    f[2 * p + 1] = row[k + 1];
  }
  return f;
}

// B-fragment: 32x16 f16. Contiguous K order (matches A frags): lane<16
// holds column N=lane, K=0..15 across the 8 VGPRs; lanes 16-31 hold K=16..31.
// With source stored K-contiguous per column (Bt[n][k], ld=32), the fragment
// is 16 contiguous halves at Bt + n*32 + (lane>>4)*16.

// ---------------------------------------------------------------------------
// f32 -> f16 conversion (grid-stride)
// ---------------------------------------------------------------------------
__global__ void f32_to_f16_kernel(const float* __restrict__ in,
                                  half_t* __restrict__ out, size_t n) {
  size_t i = (size_t)blockIdx.x * blockDim.x + threadIdx.x;
  size_t stride = (size_t)gridDim.x * blockDim.x;
  for (; i < n; i += stride) out[i] = (half_t)in[i];
}

// ---------------------------------------------------------------------------
// Tiled WMMA GEMM: C[M,N] = A[M,K] @ B[K,N] + bias[N]  (optional ReLU)
// A, B are f16 row-major. Writes f32 (outF) and/or f16 (outH) if non-null.
// Block = 256 threads (8 waves), 64x64 output tile, K-step 32.
// Wave w: rows (w&3)*16, cols (w>>2)*32 -> two 16x16 C tiles sharing one A frag.
// A tile staged via async global->LDS; B tile transposed through LDS.
// ---------------------------------------------------------------------------
__global__ __launch_bounds__(256) void gemm_f16_kernel(
    const half_t* __restrict__ A, const half_t* __restrict__ B,
    const float* __restrict__ bias, float* outF, half_t* outH,
    int M, int N, int K, int relu) {
  __shared__ half_t sA[64 * 32];   // row-major [64][32]
  __shared__ half_t sBt[64 * 32];  // transposed: sBt[n][k], [64][32]

  const int tid  = threadIdx.x;
  const int lane = tid & 31;
  const int w    = tid >> 5;
  const int tr   = w & 3;    // row-tile within block
  const int tc   = w >> 2;   // col-pair within block (0..1)

  const int rowbase = blockIdx.y * 64;
  const int colbase = blockIdx.x * 64;

  v8f c0 = {0.f, 0.f, 0.f, 0.f, 0.f, 0.f, 0.f, 0.f};
  v8f c1 = {0.f, 0.f, 0.f, 0.f, 0.f, 0.f, 0.f, 0.f};

  // Per-thread staging coordinates
  const int ar = tid >> 2;            // 0..63  (A row)
  const int ac = (tid & 3) * 8;       // 0/8/16/24 (A col, 16B chunk)
  const int bk = tid >> 3;            // 0..31  (B k)
  const int bn = (tid & 7) * 8;       // 0..56  (B n)

  for (int k0 = 0; k0 < K; k0 += 32) {
    // Stage A tile 64x32: one async 16B copy per thread
    stage16B(A + (size_t)(rowbase + ar) * K + k0 + ac, sA + ar * 32 + ac);
    // Stage B tile 32x64 transposed into sBt[n][k] (coalesced 16B global read)
    {
      const half_t* src = B + (size_t)(k0 + bk) * N + colbase + bn;
#pragma unroll
      for (int i = 0; i < 8; ++i) sBt[(bn + i) * 32 + bk] = src[i];
    }
    wait_async();
    __syncthreads();

    v16h a  = load_a_frag16(sA + tr * 16 * 32, 32, lane);
    v16h b0 = *(const v16h*)(sBt + (tc * 32 + (lane & 15)) * 32 + (lane >> 4) * 16);
    v16h b1 = *(const v16h*)(sBt + (tc * 32 + 16 + (lane & 15)) * 32 + (lane >> 4) * 16);
    c0 = WMMA_F32_F16(a, b0, c0);
    c1 = WMMA_F32_F16(a, b1, c1);
    __syncthreads();
  }

  // Epilogue: C/D layout -> lane gives N=lane&15; VGPR v gives M=v+8*(lane>>4)
  const int n  = lane & 15;
  const int mh = (lane >> 4) * 8;
#pragma unroll
  for (int v = 0; v < 8; ++v) {
    const int gm  = rowbase + tr * 16 + mh + v;
    const int gn0 = colbase + tc * 32 + n;
    const int gn1 = gn0 + 16;
    float x0 = c0[v] + bias[gn0];
    float x1 = c1[v] + bias[gn1];
    if (relu) { x0 = fmaxf(x0, 0.f); x1 = fmaxf(x1, 0.f); }
    if (outF) {
      outF[(size_t)gm * N + gn0] = x0;
      outF[(size_t)gm * N + gn1] = x1;
    }
    if (outH) {
      outH[(size_t)gm * N + gn0] = (half_t)x0;
      outH[(size_t)gm * N + gn1] = (half_t)x1;
    }
  }
}

// ---------------------------------------------------------------------------
// Fused attention: per (b, h, 16-query tile): scores = Q K^T / 8, softmax,
// write f32 probs to global, ctx = P V (f16 out). 128 threads = 4 waves.
// ---------------------------------------------------------------------------
__global__ __launch_bounds__(128) void attn_kernel(
    const half_t* __restrict__ Qh, const half_t* __restrict__ Kh,
    const half_t* __restrict__ Vh, float* __restrict__ probsF,
    half_t* __restrict__ ctxH) {
  __shared__ half_t sP[16 * SEQ];       // scores -> exp -> probs (f16)
  __shared__ half_t sVt[HDIM * 32];     // V tile transposed: sVt[d][kk]
  __shared__ float  red[16][8];
  __shared__ float  rowmax[16];
  __shared__ float  rowsum[16];

  const int qt = blockIdx.x;       // 0..63
  const int h  = blockIdx.y;       // 0..15
  const int b  = blockIdx.z;       // 0..1
  const int q0 = qt * 16;

  const int tid  = threadIdx.x;
  const int lane = tid & 31;
  const int w    = tid >> 5;       // 0..3

  const size_t qkbase = ((size_t)(b * SEQ + q0)) * HID_ + (size_t)h * HDIM;
  const size_t kvrow0 = (size_t)(b * SEQ) * HID_ + (size_t)h * HDIM;

  // --- Phase 1: energy = Q K^T / 8 -------------------------------------
  v16h qa0 = load_a_frag16(Qh + qkbase, HID_, lane);
  v16h qa1 = load_a_frag16(Qh + qkbase + 32, HID_, lane);

  const int n  = lane & 15;
  const int mh = (lane >> 4) * 8;
  for (int j = 0; j < 16; ++j) {
    const int k0 = (w * 16 + j) * 16;                // key tile start
    const half_t* Kb = Kh + kvrow0 + (size_t)k0 * HID_;
    // B operand is K^T: frag = 16 contiguous halves of a K row
    v16h kb0 = *(const v16h*)(Kb + (size_t)n * HID_ + (lane >> 4) * 16);
    v16h kb1 = *(const v16h*)(Kb + (size_t)n * HID_ + 32 + (lane >> 4) * 16);
    v8f c = {0.f, 0.f, 0.f, 0.f, 0.f, 0.f, 0.f, 0.f};
    c = WMMA_F32_F16(qa0, kb0, c);
    c = WMMA_F32_F16(qa1, kb1, c);
#pragma unroll
    for (int v = 0; v < 8; ++v)
      sP[(mh + v) * SEQ + k0 + n] = (half_t)(c[v] * 0.125f);
  }
  __syncthreads();

  // --- Phase 2: softmax over keys (8 threads per query row) ------------
  {
    const int row = tid >> 3;
    const int sub = tid & 7;
    const int c0 = sub * (SEQ / 8);
    float mx = -3.0e38f;
    for (int c = 0; c < SEQ / 8; ++c)
      mx = fmaxf(mx, (float)sP[row * SEQ + c0 + c]);
    red[row][sub] = mx;
    __syncthreads();
    if (sub == 0) {
      float m = red[row][0];
#pragma unroll
      for (int i = 1; i < 8; ++i) m = fmaxf(m, red[row][i]);
      rowmax[row] = m;
    }
    __syncthreads();
    const float m = rowmax[row];
    float sum = 0.f;
    for (int c = 0; c < SEQ / 8; ++c) {
      float e = __expf((float)sP[row * SEQ + c0 + c] - m);
      sum += e;
      sP[row * SEQ + c0 + c] = (half_t)e;
    }
    red[row][sub] = sum;
    __syncthreads();
    if (sub == 0) {
      float s = 0.f;
#pragma unroll
      for (int i = 0; i < 8; ++i) s += red[row][i];
      rowsum[row] = s;
    }
    __syncthreads();
    const float inv = 1.0f / rowsum[row];
    float* gp = probsF + (((size_t)(b * HEADS + h) * SEQ) + q0 + row) * SEQ;
    for (int c = 0; c < SEQ / 8; ++c) {
      float p = (float)sP[row * SEQ + c0 + c] * inv;
      sP[row * SEQ + c0 + c] = (half_t)p;
      gp[c0 + c] = p;   // f32 attention output
    }
  }
  __syncthreads();

  // --- Phase 3: ctx = P @ V (wave w owns d-cols w*16..w*16+15) ----------
  v8f acc = {0.f, 0.f, 0.f, 0.f, 0.f, 0.f, 0.f, 0.f};
  const int r  = lane & 15;
  const int hh = lane >> 4;
  for (int step = 0; step < 32; ++step) {
    const int kb = step * 32;
    // stage V tile 32(k) x 64(d) transposed -> sVt[d][kk]
    {
      const int kk = tid >> 2;             // 0..31
      const int dd = (tid & 3) * 16;       // 0,16,32,48
      const half_t* src = Vh + kvrow0 + (size_t)(kb + kk) * HID_ + dd;
#pragma unroll
      for (int i = 0; i < 16; ++i) sVt[(dd + i) * 32 + kk] = src[i];
    }
    __syncthreads();
    // A frag from probs (f16 in LDS)
    v16h pa;
#pragma unroll
    for (int p = 0; p < 8; ++p) {
      int k = kb + hh * 16 + 2 * p;
      pa[2 * p]     = sP[r * SEQ + k];
      pa[2 * p + 1] = sP[r * SEQ + k + 1];
    }
    v16h vb = *(const v16h*)(sVt + (w * 16 + (lane & 15)) * 32 + hh * 16);
    acc = WMMA_F32_F16(pa, vb, acc);
    __syncthreads();
  }
#pragma unroll
  for (int v = 0; v < 8; ++v) {
    const int m   = mh + v;
    const size_t row = (size_t)(b * SEQ + q0 + m) * HID_;
    const int col = h * HDIM + w * 16 + n;
    ctxH[row + col] = (half_t)acc[v];
  }
}

// ---------------------------------------------------------------------------
// y = LayerNorm(a + r) * g + beta ; writes f32 (outF) and optional f16 (outH)
// One 256-thread block per row of 1024. Population variance (jnp.var).
// ---------------------------------------------------------------------------
__global__ __launch_bounds__(256) void add_ln_kernel(
    const float* __restrict__ a, const float* __restrict__ r,
    const float* __restrict__ g, const float* __restrict__ beta,
    float* __restrict__ outF, half_t* outH) {
  __shared__ float red[256];
  const int row = blockIdx.x;
  const int tid = threadIdx.x;
  const size_t base = (size_t)row * HID_;

  float x[4];
  float s = 0.f;
#pragma unroll
  for (int i = 0; i < 4; ++i) {
    int c = tid + i * 256;
    x[i] = a[base + c] + r[base + c];
    s += x[i];
  }
  red[tid] = s;
  __syncthreads();
  for (int st = 128; st > 0; st >>= 1) {
    if (tid < st) red[tid] += red[tid + st];
    __syncthreads();
  }
  const float mu = red[0] * (1.0f / HID_);
  __syncthreads();

  float sq = 0.f;
#pragma unroll
  for (int i = 0; i < 4; ++i) {
    float d = x[i] - mu;
    sq += d * d;
  }
  red[tid] = sq;
  __syncthreads();
  for (int st = 128; st > 0; st >>= 1) {
    if (tid < st) red[tid] += red[tid + st];
    __syncthreads();
  }
  const float var = red[0] * (1.0f / HID_);
  const float rs = rsqrtf(var + 1e-5f);

#pragma unroll
  for (int i = 0; i < 4; ++i) {
    int c = tid + i * 256;
    float y = (x[i] - mu) * rs * g[c] + beta[c];
    outF[base + c] = y;
    if (outH) outH[base + c] = (half_t)y;
  }
}

// ---------------------------------------------------------------------------
// Host launcher
// ---------------------------------------------------------------------------
extern "C" void kernel_launch(void* const* d_in, const int* in_sizes, int n_in,
                              void* d_out, int out_size, void* d_ws, size_t ws_size,
                              hipStream_t stream) {
  (void)in_sizes; (void)n_in; (void)out_size; (void)ws_size;

  const float* src   = (const float*)d_in[0];
  // d_in[1] pos_embed, d_in[2] x: unused by reference
  const float* Wq    = (const float*)d_in[3];
  const float* bq    = (const float*)d_in[4];
  const float* Wk    = (const float*)d_in[5];
  const float* bk    = (const float*)d_in[6];
  const float* Wv    = (const float*)d_in[7];
  const float* bv    = (const float*)d_in[8];
  const float* Wo    = (const float*)d_in[9];
  const float* bo    = (const float*)d_in[10];
  const float* ln1_g = (const float*)d_in[11];
  const float* ln1_b = (const float*)d_in[12];
  const float* W1    = (const float*)d_in[13];
  const float* b1    = (const float*)d_in[14];
  const float* W2    = (const float*)d_in[15];
  const float* b2    = (const float*)d_in[16];
  const float* ln2_g = (const float*)d_in[17];
  const float* ln2_b = (const float*)d_in[18];

  float* outF   = (float*)d_out;                       // [2,1024,1024]
  float* probsF = (float*)d_out + (size_t)ROWS * HID_; // [2,16,1024,1024]

  // Workspace carve-up (all sizes are multiples of 256B)
  char* ws = (char*)d_ws;
  size_t off = 0;
  auto alloc = [&](size_t bytes) { char* p = ws + off; off += bytes; return p; };
  const size_t MH   = (size_t)ROWS * HID_;   // 2048*1024
  half_t* srcH = (half_t*)alloc(MH * 2);
  half_t* WqH  = (half_t*)alloc((size_t)HID_ * HID_ * 2);
  half_t* WkH  = (half_t*)alloc((size_t)HID_ * HID_ * 2);
  half_t* WvH  = (half_t*)alloc((size_t)HID_ * HID_ * 2);
  half_t* WoH  = (half_t*)alloc((size_t)HID_ * HID_ * 2);
  half_t* W1H  = (half_t*)alloc((size_t)HID_ * PF_ * 2);
  half_t* W2H  = (half_t*)alloc((size_t)PF_ * HID_ * 2);
  half_t* Qh   = (half_t*)alloc(MH * 2);
  half_t* Kh   = (half_t*)alloc(MH * 2);
  half_t* Vh   = (half_t*)alloc(MH * 2);
  half_t* ctxH = (half_t*)alloc(MH * 2);
  float*  attnOutF = (float*)alloc(MH * 4);
  float*  hF   = (float*)alloc(MH * 4);
  half_t* hH   = (half_t*)alloc(MH * 2);
  half_t* ff1H = (half_t*)alloc((size_t)ROWS * PF_ * 2);
  float*  ff2F = (float*)alloc(MH * 4);

  // 1) f32 -> f16 conversions
  auto cvt = [&](const float* in, half_t* out, size_t nElem) {
    int blocks = (int)((nElem + 255) / 256);
    if (blocks > 4096) blocks = 4096;
    f32_to_f16_kernel<<<blocks, 256, 0, stream>>>(in, out, nElem);
  };
  cvt(src, srcH, MH);
  cvt(Wq, WqH, (size_t)HID_ * HID_);
  cvt(Wk, WkH, (size_t)HID_ * HID_);
  cvt(Wv, WvH, (size_t)HID_ * HID_);
  cvt(Wo, WoH, (size_t)HID_ * HID_);
  cvt(W1, W1H, (size_t)HID_ * PF_);
  cvt(W2, W2H, (size_t)PF_ * HID_);

  // 2) Q/K/V projections: [2048,1024] @ [1024,1024] + bias -> f16
  dim3 gProj(HID_ / 64, ROWS / 64);
  gemm_f16_kernel<<<gProj, 256, 0, stream>>>(srcH, WqH, bq, nullptr, Qh,
                                             ROWS, HID_, HID_, 0);
  gemm_f16_kernel<<<gProj, 256, 0, stream>>>(srcH, WkH, bk, nullptr, Kh,
                                             ROWS, HID_, HID_, 0);
  gemm_f16_kernel<<<gProj, 256, 0, stream>>>(srcH, WvH, bv, nullptr, Vh,
                                             ROWS, HID_, HID_, 0);

  // 3) Attention: probs (f32, to d_out) + ctx (f16)
  dim3 gAttn(SEQ / 16, HEADS, BATCH);
  attn_kernel<<<gAttn, 128, 0, stream>>>(Qh, Kh, Vh, probsF, ctxH);

  // 4) Output projection: ctx @ Wo + bo -> f32
  gemm_f16_kernel<<<gProj, 256, 0, stream>>>(ctxH, WoH, bo, attnOutF, nullptr,
                                             ROWS, HID_, HID_, 0);

  // 5) h = LN(src + attn_out)
  add_ln_kernel<<<ROWS, 256, 0, stream>>>(src, attnOutF, ln1_g, ln1_b, hF, hH);

  // 6) ff1 = relu(h @ W1 + b1) -> f16
  dim3 gFF1(PF_ / 64, ROWS / 64);
  gemm_f16_kernel<<<gFF1, 256, 0, stream>>>(hH, W1H, b1, nullptr, ff1H,
                                            ROWS, PF_, HID_, 1);

  // 7) ff2 = ff1 @ W2 + b2 -> f32
  gemm_f16_kernel<<<gProj, 256, 0, stream>>>(ff1H, W2H, b2, ff2F, nullptr,
                                             ROWS, HID_, PF_, 0);

  // 8) out = LN(h + ff2) -> d_out
  add_ln_kernel<<<ROWS, 256, 0, stream>>>(hF, ff2F, ln2_g, ln2_b, outF, nullptr);
}
